// LSTMBaselineModel_36876589203492
// MI455X (gfx1250) — matrix operand order repaired and checked by reading, not compiled, not run on hardware
//
#include <hip/hip_runtime.h>

// ---------------------------------------------------------------------------
// LSTM (SEQ=256, B=512, IN=300, H=512) + BN -> FC(512->256) -> BN -> ReLU ->
// FC(256->1) -> 4*sigmoid, for MI455X (gfx1250, wave32, WMMA).
//
// Round 2 changes:
//  - k_lstm_step: M=32 per wave (4 gates x 2 M-tiles = 8 f32 accumulators),
//    halving B-operand L2 refetch per output element.
//  - Block-shared A slab (32 x 832 bf16 = 52 KB) staged into LDS with
//    GLOBAL_LOAD_ASYNC_TO_LDS_B128 (ASYNCcnt) + s_wait_asynccnt + barrier;
//    K-loop A operands then come from LDS (ds_load), cutting A L2 traffic 8x.
// ---------------------------------------------------------------------------

typedef __attribute__((ext_vector_type(16))) __bf16 v16bf;
typedef __attribute__((ext_vector_type(8)))  __bf16 v8bf;
typedef __attribute__((ext_vector_type(8)))  float  v8f;

#define SEQ    256
#define BATCH  512
#define IN_DIM 300
#define H_DIM  512
#define L_DIM  256
#define G_DIM  2048          // 4*H
#define KP     832           // padded K (300 + 512 -> 26*32)
#define BN_EPS 1e-5f

__device__ __forceinline__ float sigf(float x) {
    return 1.0f / (1.0f + __expf(-x));
}
__device__ __forceinline__ float tanh_fast(float x) {
    // safe at +/-inf: 1 - 2/(e^{2x}+1)
    return 1.0f - 2.0f / (__expf(2.0f * x) + 1.0f);
}

// ------------------------- setup kernels -----------------------------------

__global__ void k_pack_B(const float* __restrict__ W_ih,
                         const float* __restrict__ W_hh,
                         const float* __restrict__ b_ih,
                         const float* __restrict__ b_hh,
                         __bf16* __restrict__ Bp,
                         float* __restrict__ bias) {
    int stride = gridDim.x * blockDim.x;
    int idx = blockIdx.x * blockDim.x + threadIdx.x;
    const int total = G_DIM * KP;
    for (int i = idx; i < total; i += stride) {
        int g = i / KP, k = i - g * KP;
        float v = 0.0f;
        if (k < IN_DIM)              v = W_ih[g * IN_DIM + k];
        else if (k < IN_DIM + H_DIM) v = W_hh[g * H_DIM + (k - IN_DIM)];
        Bp[i] = (__bf16)v;
    }
    for (int g = idx; g < G_DIM; g += stride)
        bias[g] = b_ih[g] + b_hh[g];
}

__global__ void k_pack_W1(const float* __restrict__ W1, __bf16* __restrict__ W1p) {
    int i = blockIdx.x * blockDim.x + threadIdx.x;      // L_DIM*H_DIM threads
    if (i < L_DIM * H_DIM) W1p[i] = (__bf16)W1[i];
}

__global__ void k_zero_hc(float* __restrict__ h, float* __restrict__ c) {
    int i = blockIdx.x * blockDim.x + threadIdx.x;      // BATCH*H_DIM threads
    if (i < BATCH * H_DIM) { h[i] = 0.0f; c[i] = 0.0f; }
}

// ------------------------- per-step kernels --------------------------------

__global__ void k_pack_A(const float* __restrict__ x,
                         const float* __restrict__ h,
                         __bf16* __restrict__ Ap, int t) {
    int i = blockIdx.x * blockDim.x + threadIdx.x;      // BATCH*KP threads
    if (i >= BATCH * KP) return;
    int b = i / KP, k = i - b * KP;
    float v = 0.0f;
    if (k < IN_DIM)              v = x[(t * BATCH + b) * IN_DIM + k];
    else if (k < IN_DIM + H_DIM) v = h[b * H_DIM + (k - IN_DIM)];
    Ap[i] = (__bf16)v;
}

// A-operand (16-bit A 16x32 layout): lane L (r=L&15, half=L>>4) row r holds
//   elems 0..7  -> K = k0 + half*8 + i
//   elems 8..15 -> K = k0 + 16 + half*8 + i
__device__ __forceinline__ v16bf load_a_tile(const __bf16* __restrict__ row,
                                             int k0, int half) {
    v8bf lo = *(const v8bf*)(row + k0 + half * 8);
    v8bf hi = *(const v8bf*)(row + k0 + 16 + half * 8);
    return __builtin_shufflevector(lo, hi, 0,1,2,3,4,5,6,7,8,9,10,11,12,13,14,15);
}

// B-operand (16-bit B 32x16 layout): lane L holds column n=L&15;
//   elems 0..7 -> K = k0 + half*8 + i ; elems 8..15 -> K = k0 + 16 + half*8 + i.
__device__ __forceinline__ v16bf load_b_tile(const __bf16* __restrict__ col,
                                             int k0, int half) {
    v8bf lo = *(const v8bf*)(col + k0 + half * 8);
    v8bf hi = *(const v8bf*)(col + k0 + 16 + half * 8);
    return __builtin_shufflevector(lo, hi, 0,1,2,3,4,5,6,7,8,9,10,11,12,13,14,15);
}

#define WMMA_BF16(A, B, C) \
    __builtin_amdgcn_wmma_f32_16x16x32_bf16(false, (A), false, (B), (short)0, (C), false, false)

// One step: gates = [x_t | h] @ [W_ih | W_hh]^T + bias, then cell update.
// Per wave: 32 (batch) x 16 (hidden) output, 4 gate accumulators x 2 M-tiles.
// Block (8 waves) shares one 32-row A slab, staged to LDS via async copies.
__global__ void __launch_bounds__(256)
k_lstm_step(const __bf16* __restrict__ Ap,
            const __bf16* __restrict__ Bp,
            const float*  __restrict__ bias,
            float* __restrict__ h, float* __restrict__ c) {
    __shared__ __bf16 Asl[32 * KP];            // 53,248 B

    const int wave = threadIdx.x >> 5;
    const int lane = threadIdx.x & 31;
    const int mg = blockIdx.x >> 2;            // M-group 0..15 (32 batch rows)
    const int ht = (blockIdx.x & 3) * 8 + wave; // hidden tile 0..31
    const int r = lane & 15;
    const int half = lane >> 4;

    // ---- async-stage the block's A slab: 32 x 832 bf16 -> LDS -------------
    {
        const unsigned long long gbase =
            (unsigned long long)(const char*)(Ap + (size_t)mg * 32 * KP);
        const unsigned lds_base = (unsigned)(unsigned long long)(&Asl[0]);
        #pragma unroll
        for (int i = 0; i < 13; ++i) {          // 13 * 256 * 16B = 53,248 B
            const int chunk = i * 256 + threadIdx.x;
            unsigned lds_addr = lds_base + chunk * 16;
            unsigned long long gaddr = gbase + (unsigned long long)chunk * 16;
            asm volatile("global_load_async_to_lds_b128 %0, %1, off"
                         :: "v"(lds_addr), "v"(gaddr) : "memory");
        }
        asm volatile("s_wait_asynccnt 0x0" ::: "memory");
        __syncthreads();
    }

    const int n = ht * 16 + r;                 // hidden column for this lane
    const __bf16* br0 = Bp + (size_t)(0 * H_DIM + n) * KP;
    const __bf16* br1 = Bp + (size_t)(1 * H_DIM + n) * KP;
    const __bf16* br2 = Bp + (size_t)(2 * H_DIM + n) * KP;
    const __bf16* br3 = Bp + (size_t)(3 * H_DIM + n) * KP;

    const __bf16* arow0 = &Asl[(size_t)r * KP];        // M-tile 0 rows 0..15
    const __bf16* arow1 = &Asl[(size_t)(16 + r) * KP]; // M-tile 1 rows 16..31

    v8f acc[4][2];
    #pragma unroll
    for (int q = 0; q < 4; ++q) { acc[q][0] = (v8f){}; acc[q][1] = (v8f){}; }

    for (int k0 = 0; k0 < KP; k0 += 32) {
        v16bf a0 = load_a_tile(arow0, k0, half);       // ds_load from LDS
        v16bf a1 = load_a_tile(arow1, k0, half);
        v16bf b0 = load_b_tile(br0, k0, half);
        acc[0][0] = WMMA_BF16(a0, b0, acc[0][0]);
        acc[0][1] = WMMA_BF16(a1, b0, acc[0][1]);
        v16bf b1 = load_b_tile(br1, k0, half);
        acc[1][0] = WMMA_BF16(a0, b1, acc[1][0]);
        acc[1][1] = WMMA_BF16(a1, b1, acc[1][1]);
        v16bf b2 = load_b_tile(br2, k0, half);
        acc[2][0] = WMMA_BF16(a0, b2, acc[2][0]);
        acc[2][1] = WMMA_BF16(a1, b2, acc[2][1]);
        v16bf b3 = load_b_tile(br3, k0, half);
        acc[3][0] = WMMA_BF16(a0, b3, acc[3][0]);
        acc[3][1] = WMMA_BF16(a1, b3, acc[3][1]);
    }

    const float bi = bias[n];
    const float bf = bias[H_DIM + n];
    const float bg = bias[2 * H_DIM + n];
    const float bo = bias[3 * H_DIM + n];

    // C/D tile layout: VGPR v, lanes 0-15 -> M=v, lanes 16-31 -> M=v+8, N=r.
    #pragma unroll
    for (int mt = 0; mt < 2; ++mt) {
        #pragma unroll
        for (int v = 0; v < 8; ++v) {
            int row = mg * 32 + mt * 16 + v + half * 8;
            size_t idx = (size_t)row * H_DIM + n;
            float iv = sigf(acc[0][mt][v] + bi);
            float fv = sigf(acc[1][mt][v] + bf);
            float gv = tanh_fast(acc[2][mt][v] + bg);
            float ov = sigf(acc[3][mt][v] + bo);
            float cn = fv * c[idx] + iv * gv;
            c[idx] = cn;
            h[idx] = ov * tanh_fast(cn);
        }
    }
}

// ------------------------- tail kernels ------------------------------------

// Per-column batch stats; fold BN affine into scale a[] and shift d[].
__global__ void k_bn_stats(const float* __restrict__ m, int ncols, int nrows,
                           const float* __restrict__ gamma,
                           const float* __restrict__ beta,
                           float* __restrict__ a, float* __restrict__ d) {
    int col = blockIdx.x * blockDim.x + threadIdx.x;
    if (col >= ncols) return;
    float s = 0.0f, sq = 0.0f;
    for (int rdx = 0; rdx < nrows; ++rdx) {
        float v = m[(size_t)rdx * ncols + col];
        s += v; sq += v * v;
    }
    float inv_n = 1.0f / (float)nrows;
    float mean = s * inv_n;
    float var = sq * inv_n - mean * mean;
    float scale = rsqrtf(var + BN_EPS) * gamma[col];
    a[col] = scale;
    d[col] = beta[col] - mean * scale;
}

__global__ void k_pack_A2(const float* __restrict__ h,
                          const float* __restrict__ a1,
                          const float* __restrict__ d1,
                          __bf16* __restrict__ A2) {
    int i = blockIdx.x * blockDim.x + threadIdx.x;      // BATCH*H_DIM threads
    if (i >= BATCH * H_DIM) return;
    int col = i & (H_DIM - 1);
    A2[i] = (__bf16)(h[i] * a1[col] + d1[col]);
}

__global__ void __launch_bounds__(256)
k_gemm2(const __bf16* __restrict__ A2, const __bf16* __restrict__ W1p,
        const float* __restrict__ b1, float* __restrict__ y) {
    const int wave = threadIdx.x >> 5;
    const int lane = threadIdx.x & 31;
    const int tile = blockIdx.x * 8 + wave;   // 0..511
    const int bt = tile >> 4;                 // batch tile 0..31
    const int lt = tile & 15;                 // L tile     0..15
    const int r = lane & 15;
    const int half = lane >> 4;

    const __bf16* arow = A2 + (size_t)(bt * 16 + r) * H_DIM;
    const __bf16* brow = W1p + (size_t)(lt * 16 + r) * H_DIM;

    v8f acc = {};
    for (int k0 = 0; k0 < H_DIM; k0 += 32) {
        v16bf a = load_a_tile(arow, k0, half);
        v16bf b = load_b_tile(brow, k0, half);
        acc = WMMA_BF16(a, b, acc);
    }
    const int l = lt * 16 + r;
    const float bb = b1[l];
    #pragma unroll
    for (int v = 0; v < 8; ++v) {
        int row = bt * 16 + v + half * 8;
        y[(size_t)row * L_DIM + l] = acc[v] + bb;
    }
}

__global__ void k_final(const float* __restrict__ y,
                        const float* __restrict__ a2,
                        const float* __restrict__ d2,
                        const float* __restrict__ W2,
                        const float* __restrict__ b2,
                        float* __restrict__ out) {
    int b = blockIdx.x * blockDim.x + threadIdx.x;
    if (b >= BATCH) return;
    float acc = b2[0];
    for (int l = 0; l < L_DIM; ++l) {
        float v = y[(size_t)b * L_DIM + l] * a2[l] + d2[l];
        v = v > 0.0f ? v : 0.0f;               // ReLU
        acc += v * W2[l];
    }
    out[b] = 4.0f * sigf(acc);
}

// ------------------------- launcher ----------------------------------------

extern "C" void kernel_launch(void* const* d_in, const int* in_sizes, int n_in,
                              void* d_out, int out_size, void* d_ws, size_t ws_size,
                              hipStream_t stream) {
    const float* x    = (const float*)d_in[0];
    const float* W_ih = (const float*)d_in[1];
    const float* W_hh = (const float*)d_in[2];
    const float* b_ih = (const float*)d_in[3];
    const float* b_hh = (const float*)d_in[4];
    const float* bn1g = (const float*)d_in[5];
    const float* bn1b = (const float*)d_in[6];
    const float* W1   = (const float*)d_in[7];
    const float* b1   = (const float*)d_in[8];
    const float* bn2g = (const float*)d_in[9];
    const float* bn2b = (const float*)d_in[10];
    const float* W2   = (const float*)d_in[11];
    const float* b2   = (const float*)d_in[12];
    float* out = (float*)d_out;

    // workspace layout (~7.7 MB, all offsets 1 KB-aligned)
    char* w = (char*)d_ws;
    __bf16* Bp  = (__bf16*)w; w += (size_t)G_DIM * KP * 2;     // 3,407,872
    __bf16* Ap  = (__bf16*)w; w += (size_t)BATCH * KP * 2;     //   851,968
    __bf16* W1p = (__bf16*)w; w += (size_t)L_DIM * H_DIM * 2;  //   262,144
    __bf16* A2  = (__bf16*)w; w += (size_t)BATCH * H_DIM * 2;  //   524,288
    float* h    = (float*)w;  w += (size_t)BATCH * H_DIM * 4;  // 1,048,576
    float* c    = (float*)w;  w += (size_t)BATCH * H_DIM * 4;  // 1,048,576
    float* bias = (float*)w;  w += (size_t)G_DIM * 4;          //     8,192
    float* y    = (float*)w;  w += (size_t)BATCH * L_DIM * 4;  //   524,288
    float* a1   = (float*)w;  w += (size_t)H_DIM * 4;
    float* d1   = (float*)w;  w += (size_t)H_DIM * 4;
    float* a2   = (float*)w;  w += (size_t)L_DIM * 4;
    float* d2   = (float*)w;  w += (size_t)L_DIM * 4;

    // setup: pack weights to bf16, zero recurrent state (every call)
    k_pack_B<<<1024, 256, 0, stream>>>(W_ih, W_hh, b_ih, b_hh, Bp, bias);
    k_pack_W1<<<(L_DIM * H_DIM + 255) / 256, 256, 0, stream>>>(W1, W1p);
    k_zero_hc<<<(BATCH * H_DIM + 255) / 256, 256, 0, stream>>>(h, c);

    // recurrence: 256 sequential steps
    const int packA_blocks = (BATCH * KP + 255) / 256;
    for (int t = 0; t < SEQ; ++t) {
        k_pack_A<<<packA_blocks, 256, 0, stream>>>(x, h, Ap, t);
        k_lstm_step<<<64, 256, 0, stream>>>(Ap, Bp, bias, h, c);
    }

    // BN1 -> FC1 -> BN2 -> ReLU -> FC2 -> 4*sigmoid
    k_bn_stats<<<(H_DIM + 255) / 256, 256, 0, stream>>>(h, H_DIM, BATCH,
                                                        bn1g, bn1b, a1, d1);
    k_pack_A2<<<(BATCH * H_DIM + 255) / 256, 256, 0, stream>>>(h, a1, d1, A2);
    k_gemm2<<<64, 256, 0, stream>>>(A2, W1p, b1, y);
    k_bn_stats<<<(L_DIM + 255) / 256, 256, 0, stream>>>(y, L_DIM, BATCH,
                                                        bn2g, bn2b, a2, d2);
    k_final<<<(BATCH + 255) / 256, 256, 0, stream>>>(y, a2, d2, W2, b2, out);
}
